// LogicLayer_41910290874770
// MI455X (gfx1250) — matrix operand order repaired and checked by reading, not compiled, not run on hardware
//
#include <hip/hip_runtime.h>
#include <math.h>

typedef unsigned int   u32;
typedef unsigned short u16;
typedef __attribute__((ext_vector_type(16))) __bf16 v16bf;
typedef __attribute__((ext_vector_type(8)))  float  v8f;
typedef __attribute__((ext_vector_type(4)))  u32    v4u;
typedef __attribute__((ext_vector_type(8)))  u32    v8u;

#define SIZE  1024   // M (output rows)
#define PREV  1024   // K
#define BATCH 16384  // N

// ---------- bf16 helpers (round-to-nearest-even via bit trick) ----------
__device__ inline u16 bf16_rne(float f) {
  u32 u = __float_as_uint(f);
  u32 r = u + 0x7FFFu + ((u >> 16) & 1u);
  return (u16)(r >> 16);
}
__device__ inline float bf16_f32(u16 h) { return __uint_as_float(((u32)h) << 16); }
__device__ inline void split_bf16(float f, u16& hi, u16& lo) {
  hi = bf16_rne(f);
  lo = bf16_rne(f - bf16_f32(hi));
}

// ---------- kernel 1: table softmax -> affine gate coefficients ----------
// gates are affine in {1, A, B, AB}; fold pT against constant coefficients.
__global__ __launch_bounds__(256) void table_coeffs(const float* __restrict__ T,
                                                    float* __restrict__ W) {
  int s = blockIdx.x * blockDim.x + threadIdx.x;
  if (s >= SIZE) return;
  float t[16], mx = -3.4e38f;
#pragma unroll
  for (int g = 0; g < 16; ++g) { t[g] = T[g * SIZE + s]; mx = fmaxf(mx, t[g]); }
  float sum = 0.f;
#pragma unroll
  for (int g = 0; g < 16; ++g) { t[g] = expf(t[g] - mx); sum += t[g]; }
  float inv = 1.f / sum;
  const float c0[16]  = {0,0,0,0, 0,0,0,0, 1, 1, 1, 1, 1, 1, 1,1};
  const float cA[16]  = {0,0,1,1, 0,0,1,1,-1,-1, 0, 0,-1,-1, 0,0};
  const float cB[16]  = {0,0,0,0, 1,1,1,1,-1,-1,-1,-1, 0, 0, 0,0};
  const float cAB[16] = {0,1,-1,0,-1,0,-2,-1, 1, 2, 0, 1, 0, 1,-1,0};
  float w0 = 0, w1 = 0, w2 = 0, w3 = 0;
#pragma unroll
  for (int g = 0; g < 16; ++g) {
    float p = t[g] * inv;
    w0 += c0[g] * p; w1 += cA[g] * p; w2 += cB[g] * p; w3 += cAB[g] * p;
  }
  W[s] = w0; W[SIZE + s] = w1; W[2 * SIZE + s] = w2; W[3 * SIZE + s] = w3;
}

// ---------- kernel 2: row softmax of WA/WB -> bf16 hi/lo splits ----------
__global__ __launch_bounds__(256) void softmax_split_rows(
    const float* __restrict__ WA, const float* __restrict__ WB,
    u16* __restrict__ Ahi, u16* __restrict__ Alo,
    u16* __restrict__ Bhi, u16* __restrict__ Blo) {
  __shared__ float red[256];
  int row = blockIdx.x;
  const float* src; u16* dh; u16* dl;
  if (row < SIZE) {
    src = WA + (size_t)row * PREV; dh = Ahi + (size_t)row * PREV; dl = Alo + (size_t)row * PREV;
  } else {
    int r = row - SIZE;
    src = WB + (size_t)r * PREV; dh = Bhi + (size_t)r * PREV; dl = Blo + (size_t)r * PREV;
  }
  int tid = threadIdx.x;
  float v[4];
  float mx = -3.4e38f;
#pragma unroll
  for (int i = 0; i < 4; ++i) { v[i] = src[tid + i * 256]; mx = fmaxf(mx, v[i]); }
  red[tid] = mx; __syncthreads();
  for (int s = 128; s > 0; s >>= 1) {
    if (tid < s) red[tid] = fmaxf(red[tid], red[tid + s]);
    __syncthreads();
  }
  mx = red[0]; __syncthreads();
  float sum = 0.f;
#pragma unroll
  for (int i = 0; i < 4; ++i) { v[i] = expf(v[i] - mx); sum += v[i]; }
  red[tid] = sum; __syncthreads();
  for (int s = 128; s > 0; s >>= 1) {
    if (tid < s) red[tid] += red[tid + s];
    __syncthreads();
  }
  float inv = 1.0f / red[0];
#pragma unroll
  for (int i = 0; i < 4; ++i) {
    u16 h, l; split_bf16(v[i] * inv, h, l);
    dh[tid + i * 256] = h; dl[tid + i * 256] = l;
  }
}

// ---------- kernel 3: transpose P (KxN f32) -> Pt (NxK) bf16 hi/lo ----------
__global__ __launch_bounds__(256) void transpose_split(const float* __restrict__ P,
                                                       u16* __restrict__ Phi,
                                                       u16* __restrict__ Plo) {
  __shared__ float tile[32][33];
  int nb = blockIdx.x * 32, kb = blockIdx.y * 32;
  int tx = threadIdx.x, ty = threadIdx.y;  // 32 x 8
#pragma unroll
  for (int i = 0; i < 4; ++i) {
    int k = kb + ty + i * 8;
    tile[ty + i * 8][tx] = P[(size_t)k * BATCH + nb + tx];
  }
  __syncthreads();
#pragma unroll
  for (int i = 0; i < 4; ++i) {
    int n = nb + ty + i * 8;
    float v = tile[tx][ty + i * 8];
    u16 h, l; split_bf16(v, h, l);
    Phi[(size_t)n * PREV + kb + tx] = h;
    Plo[(size_t)n * PREV + kb + tx] = l;
  }
}

// ---------- fragment loaders matching CDNA5 WMMA bf16 VGPR layouts ----------
// A 16x32 (MxK): lane(0-15)=row m, half-group lh=lane>>4; halves 0-7 -> K = lh*8+0..7,
// halves 8-15 -> K = 16+lh*8+0..7  => two contiguous 16B chunks at k0+lh*8 and +16.
__device__ inline v16bf frag_a(const u16* __restrict__ rowp, int off0) {
  v8u t;
  t.lo = *(const v4u*)(rowp + off0);
  t.hi = *(const v4u*)(rowp + off0 + 16);
  return __builtin_bit_cast(v16bf, t);
}
// B 32x16 (KxN): lane holds column n=lane&15; halves 0-7 -> K = lh*8+0..7, halves 8-15 -> K = 16+lh*8+0..7
// => two contiguous 16B chunks at k0 + lh*8 and +16 from the NxK-transposed matrix.
__device__ inline v16bf frag_b(const u16* __restrict__ rowp, int off0) {
  v8u t;
  t.lo = *(const v4u*)(rowp + off0);
  t.hi = *(const v4u*)(rowp + off0 + 16);
  return __builtin_bit_cast(v16bf, t);
}

// ---------- kernel 4: fused dual-GEMM (bf16x3 split) + affine gate epilogue ----------
// block = 8 waves, tile 128(M) x 64(N); wave w -> rows [w*16, w*16+16), 4 N-subtiles.
__global__ __launch_bounds__(256) void logic_gemm_wmma(
    const u16* __restrict__ Ahi, const u16* __restrict__ Alo,
    const u16* __restrict__ Bhi, const u16* __restrict__ Blo,
    const u16* __restrict__ Phi, const u16* __restrict__ Plo,
    const float* __restrict__ W, float* __restrict__ out) {
  const int K = PREV;
  int lane = threadIdx.x & 31;
  int wave = threadIdx.x >> 5;
  int l15 = lane & 15, lh = lane >> 4;
  int mbase = blockIdx.y * 128 + wave * 16;
  int nbase = blockIdx.x * 64;

  const u16* aHrow = Ahi + (size_t)(mbase + l15) * K;
  const u16* aLrow = Alo + (size_t)(mbase + l15) * K;
  const u16* bHrow = Bhi + (size_t)(mbase + l15) * K;
  const u16* bLrow = Blo + (size_t)(mbase + l15) * K;
  const u16* pHrow[4];
  const u16* pLrow[4];
#pragma unroll
  for (int s = 0; s < 4; ++s) {
    pHrow[s] = Phi + (size_t)(nbase + s * 16 + l15) * K;
    pLrow[s] = Plo + (size_t)(nbase + s * 16 + l15) * K;
  }

  v8f accA[4], accB[4];
#pragma unroll
  for (int s = 0; s < 4; ++s) { accA[s] = (v8f){}; accB[s] = (v8f){}; }

  const int offA = lh * 8;
  const int offB = lh * 8;

#pragma unroll 1
  for (int k0 = 0; k0 < K; k0 += 32) {
    v16bf fAh = frag_a(aHrow, k0 + offA);
    v16bf fAl = frag_a(aLrow, k0 + offA);
    v16bf fBh = frag_a(bHrow, k0 + offA);
    v16bf fBl = frag_a(bLrow, k0 + offA);
#pragma unroll
    for (int s = 0; s < 4; ++s) {
      v16bf fPh = frag_b(pHrow[s], k0 + offB);
      v16bf fPl = frag_b(pLrow[s], k0 + offB);
      // a ~= Ah*Ph + Ah*Pl + Al*Ph  (bf16x3, f32 accumulate)
      accA[s] = __builtin_amdgcn_wmma_f32_16x16x32_bf16(false, fAh, false, fPh, (short)0, accA[s], false, false);
      accA[s] = __builtin_amdgcn_wmma_f32_16x16x32_bf16(false, fAh, false, fPl, (short)0, accA[s], false, false);
      accA[s] = __builtin_amdgcn_wmma_f32_16x16x32_bf16(false, fAl, false, fPh, (short)0, accA[s], false, false);
      accB[s] = __builtin_amdgcn_wmma_f32_16x16x32_bf16(false, fBh, false, fPh, (short)0, accB[s], false, false);
      accB[s] = __builtin_amdgcn_wmma_f32_16x16x32_bf16(false, fBh, false, fPl, (short)0, accB[s], false, false);
      accB[s] = __builtin_amdgcn_wmma_f32_16x16x32_bf16(false, fBl, false, fPh, (short)0, accB[s], false, false);
    }
  }

  // epilogue: out = w0 + w1*a + w2*b + w3*a*b
  // C/D layout: VGPR j, lanes 0-15 -> M=j, lanes 16-31 -> M=j+8, N=lane&15.
  float w0[8], w1[8], w2[8], w3[8];
#pragma unroll
  for (int j = 0; j < 8; ++j) {
    int m = mbase + lh * 8 + j;
    w0[j] = W[m]; w1[j] = W[SIZE + m]; w2[j] = W[2 * SIZE + m]; w3[j] = W[3 * SIZE + m];
  }
#pragma unroll
  for (int s = 0; s < 4; ++s) {
    int n = nbase + s * 16 + l15;
#pragma unroll
    for (int j = 0; j < 8; ++j) {
      int m = mbase + lh * 8 + j;
      float a = accA[s][j], b = accB[s][j];
      out[(size_t)m * BATCH + n] = w0[j] + w1[j] * a + w2[j] * b + w3[j] * (a * b);
    }
  }
}

// ---------- host-side launcher ----------
extern "C" void kernel_launch(void* const* d_in, const int* in_sizes, int n_in,
                              void* d_out, int out_size, void* d_ws, size_t ws_size,
                              hipStream_t stream) {
  const float* P  = (const float*)d_in[0];  // (1024, 16384)
  const float* WA = (const float*)d_in[1];  // (1024, 1024)
  const float* WB = (const float*)d_in[2];  // (1024, 1024)
  const float* T  = (const float*)d_in[3];  // (16, 1024)
  float* out = (float*)d_out;               // (1024, 16384)

  char* ws = (char*)d_ws;
  float* Wc  = (float*)ws;                         // 4*1024 f32 = 16 KiB
  u16*   Ahi = (u16*)(ws + (16 << 10));            // each split: 2 MiB
  u16*   Alo = Ahi + (size_t)SIZE * PREV;
  u16*   Bhi = Alo + (size_t)SIZE * PREV;
  u16*   Blo = Bhi + (size_t)SIZE * PREV;
  u16*   Phi = Blo + (size_t)SIZE * PREV;          // 32 MiB
  u16*   Plo = Phi + (size_t)BATCH * PREV;         // 32 MiB

  table_coeffs<<<dim3(4), dim3(256), 0, stream>>>(T, Wc);
  softmax_split_rows<<<dim3(2 * SIZE), dim3(256), 0, stream>>>(WA, WB, Ahi, Alo, Bhi, Blo);
  transpose_split<<<dim3(BATCH / 32, PREV / 32), dim3(32, 8), 0, stream>>>(P, Phi, Plo);
  logic_gemm_wmma<<<dim3(BATCH / 64, SIZE / 128), dim3(256), 0, stream>>>(
      Ahi, Alo, Bhi, Blo, Phi, Plo, Wc, out);
}
